// Qwen2RetNetDecoderLayer_62130996904034
// MI455X (gfx1250) — compile-verified, matrix operand order checked by reading, not run on hardware
//
#include <hip/hip_runtime.h>
#include <hip/hip_bf16.h>
#include <cstdint>

#define B_ 4
#define T_ 8192
#define D_ 1024
#define H_ 8
#define DK_ 128
#define DV_ 128
#define I_ 2816
#define C_ 64
#define NC_ (T_ / C_)
#define GATE_EPS 1e-5f
#define RMS_EPS 1e-6f

typedef __attribute__((ext_vector_type(16))) __bf16 v16bf;
typedef __attribute__((ext_vector_type(8)))  float  v8f;
typedef __attribute__((ext_vector_type(8)))  unsigned int v8u;
typedef __attribute__((ext_vector_type(4)))  unsigned int u32x4;
typedef __attribute__((ext_vector_type(4)))  int i32x4;
typedef __attribute__((ext_vector_type(8)))  int i32x8;

#if defined(__has_builtin)
#if __has_builtin(__builtin_amdgcn_tensor_load_to_lds)
#define USE_TDM 1
#endif
#endif
#ifndef USE_TDM
#define USE_TDM 0
#endif

// ---------- scalar helpers ----------
__device__ __forceinline__ float bf2f(unsigned short u) {
    return __uint_as_float(((unsigned int)u) << 16);
}
__device__ __forceinline__ unsigned short f2bf(float f) {
    unsigned int u = __float_as_uint(f);
    return (unsigned short)((u + 0x7FFFu + ((u >> 16) & 1u)) >> 16);
}
__device__ __forceinline__ float silu(float x) {
    return x / (1.0f + __expf(-x));
}
__device__ __forceinline__ v8f vzero8() {
    v8f r;
#pragma unroll
    for (int i = 0; i < 8; ++i) r[i] = 0.0f;
    return r;
}
// ISA 7.12.2 16-bit A-matrix k index for reg-pair j (0..7), lane half (0/1)
__device__ __forceinline__ int kpat(int j, int half) {
    return (j < 4 ? 2 * j : 16 + 2 * (j - 4)) + 8 * half;
}
__device__ __forceinline__ v8f wmma_bf16(v16bf a, v16bf b, v8f c) {
    return __builtin_amdgcn_wmma_f32_16x16x32_bf16(false, a, false, b, (short)0, c, false, false);
}
// LDS aperture maps addr[31:0] -> LDS byte offset
__device__ __forceinline__ unsigned lds_addr32(const void* p) {
    return (unsigned)(uintptr_t)p;
}

// ---------- TDM: issue one 2-D bf16 tile load Global -> LDS (uniform args, wave-scalar) ----------
__device__ __forceinline__ void tdm_load_tile_bf16(const unsigned short* gbase,
                                                   unsigned lds_byte_addr,
                                                   int rows, int rowlen,
                                                   long long row_stride_elems,
                                                   long long tensor_rows) {
#if USE_TDM
    unsigned long long ga = (unsigned long long)(uintptr_t)gbase;
    u32x4 g0;
    g0[0] = 1u;                                               // count=1, is_restore=0, gather off
    g0[1] = lds_byte_addr;                                    // lds_addr [63:32]
    g0[2] = (unsigned)(ga & 0xFFFFFFFFull);                   // global_addr [95:64]
    g0[3] = (unsigned)((ga >> 32) & 0x01FFFFFFull) | (2u << 30); // addr[56:32], type=2
    unsigned td0 = (unsigned)row_stride_elems;                // tensor_dim0 (>= tile_dim0)
    unsigned td1 = (unsigned)tensor_rows;                     // tensor_dim1 (>= tile_dim1)
    unsigned long long st0 = (unsigned long long)row_stride_elems; // tensor_dim0_stride
    i32x8 g1;
    g1[0] = 0x00010000;                                       // wg_mask=0, data_size=1 (2B)
    g1[1] = (int)((td0 & 0xFFFFu) << 16);                     // dim0[15:0] in bits 63:48
    g1[2] = (int)(((td0 >> 16) & 0xFFFFu) | ((td1 & 0xFFFFu) << 16));
    g1[3] = (int)(((td1 >> 16) & 0xFFFFu) | (((unsigned)rowlen & 0xFFFFu) << 16)); // tile_dim0
    g1[4] = (int)((unsigned)rows & 0xFFFFu);                  // tile_dim1 (tile_dim2 = 0)
    g1[5] = (int)(st0 & 0xFFFFFFFFull);                       // stride0 low32
    g1[6] = (int)((st0 >> 32) & 0xFFFFull);                   // stride0 hi16, stride1 lo16 = 0
    g1[7] = 0;
    i32x4 z4 = {0, 0, 0, 0};
#if __clang_major__ >= 23
    i32x8 z8 = {0, 0, 0, 0, 0, 0, 0, 0};
    __builtin_amdgcn_tensor_load_to_lds(g0, g1, z4, z4, z8, 0);
#else
    __builtin_amdgcn_tensor_load_to_lds(g0, g1, z4, z4, 0);
#endif
#else
    (void)gbase; (void)lds_byte_addr; (void)rows; (void)rowlen;
    (void)row_stride_elems; (void)tensor_rows;
#endif
}

// ---------- fragment builders (LDS, bf16) ----------
__device__ __forceinline__ v16bf frag_rowmajor(const unsigned short* base, int row0, int stride,
                                               int k0, int lane) {
    int nl = lane & 15, half = lane >> 4;
    const unsigned short* p = base + (size_t)(row0 + nl) * stride + k0;
    v8u r;
#pragma unroll
    for (int j = 0; j < 8; ++j)
        r[j] = *(const unsigned int*)(p + kpat(j, half));
    return __builtin_bit_cast(v16bf, r);
}
__device__ __forceinline__ v16bf frag_rowmajor_scaled(const unsigned short* base, int row0, int stride,
                                                      int k0, int lane, float fac) {
    int nl = lane & 15, half = lane >> 4;
    const unsigned short* p = base + (size_t)(row0 + nl) * stride + k0;
    v8u r;
#pragma unroll
    for (int j = 0; j < 8; ++j) {
        unsigned int u = *(const unsigned int*)(p + kpat(j, half));
        float f0 = bf2f((unsigned short)(u & 0xffff)) * fac;
        float f1 = bf2f((unsigned short)(u >> 16)) * fac;
        r[j] = (unsigned int)f2bf(f0) | ((unsigned int)f2bf(f1) << 16);
    }
    return __builtin_bit_cast(v16bf, r);
}
__device__ __forceinline__ v16bf frag_colstride(const unsigned short* base, int n0, int stride,
                                                int k0, int lane) {
    int nl = lane & 15, half = lane >> 4;
    const unsigned short* p = base + n0 + nl;
    v8u r;
#pragma unroll
    for (int j = 0; j < 8; ++j) {
        int kk = kpat(j, half) + k0;
        unsigned int lo = p[(size_t)kk * stride];
        unsigned int hi = p[(size_t)(kk + 1) * stride];
        r[j] = (lo & 0xffffu) | (hi << 16);
    }
    return __builtin_bit_cast(v16bf, r);
}
__device__ __forceinline__ v16bf frag_kT_dec(const unsigned short* kbl, int d0, int k0,
                                             int lane, float lg) {
    int nl = lane & 15, half = lane >> 4;
    int d = d0 + nl;
    v8u r;
#pragma unroll
    for (int j = 0; j < 8; ++j) {
        int kk = kpat(j, half) + k0;
        float f0 = bf2f(kbl[(size_t)kk * 128 + d]) * __expf(lg * (float)(63 - kk));
        float f1 = bf2f(kbl[(size_t)(kk + 1) * 128 + d]) * __expf(lg * (float)(63 - (kk + 1)));
        r[j] = (unsigned int)f2bf(f0) | ((unsigned int)f2bf(f1) << 16);
    }
    return __builtin_bit_cast(v16bf, r);
}

// ---------- weight transpose + cast: fp32 [R][Cc] -> bf16 [Cc][R] ----------
__global__ void transpose_cast_kernel(const float* __restrict__ in, unsigned short* __restrict__ out,
                                      int R, int Cc) {
    size_t idx = (size_t)blockIdx.x * 256 + threadIdx.x;
    if (idx >= (size_t)R * Cc) return;
    int r = (int)(idx / Cc), cc = (int)(idx % Cc);
    out[(size_t)cc * R + r] = f2bf(in[idx]);
}

// ---------- rmsnorm: fp32 [rows][D] -> bf16 ----------
__global__ __launch_bounds__(256) void rmsnorm_kernel(const float* __restrict__ in,
                                                      const float* __restrict__ w,
                                                      unsigned short* __restrict__ out,
                                                      float eps) {
    __shared__ float red[256];
    size_t row = blockIdx.x;
    const float* x = in + row * D_;
    float s = 0.0f;
    for (int i = threadIdx.x; i < D_; i += 256) { float v = x[i]; s += v * v; }
    red[threadIdx.x] = s;
    __syncthreads();
    for (int off = 128; off > 0; off >>= 1) {
        if ((int)threadIdx.x < off) red[threadIdx.x] += red[threadIdx.x + off];
        __syncthreads();
    }
    float inv = rsqrtf(red[0] / (float)D_ + eps);
    for (int i = threadIdx.x; i < D_; i += 256)
        out[row * D_ + i] = f2bf(x[i] * inv * w[i]);
}

// ---------- WMMA GEMM: C[M,N] = A[M,K](bf16) @ Bt[N,K]^T(bf16) ----------
// EPI: 0 = store bf16, 1 = store f32, 2 = store f32 + residual add
#define BM 128
#define BN 64
#define BK 32
template <int EPI>
__global__ __launch_bounds__(256) void gemm_bf16(const unsigned short* __restrict__ A,
                                                 const unsigned short* __restrict__ Bt,
                                                 void* __restrict__ Cout,
                                                 const float* __restrict__ Res,
                                                 int M, int N, int K) {
    __shared__ unsigned short As[2][BM * BK];  // 2 x 8 KB
    __shared__ unsigned short Bs[2][BN * BK];  // 2 x 4 KB
    int tid = threadIdx.x;
    int lane = tid & 31, wave = tid >> 5;
    int bm = blockIdx.y * BM;
    int bn = blockIdx.x * BN;
    v8f acc[4];
#pragma unroll
    for (int t = 0; t < 4; ++t) acc[t] = vzero8();
    int niter = K / BK;

#if USE_TDM
    // TDM double-buffered pipeline: wave 0 drives DMA, all waves run WMMA.
    if (wave == 0) {
        tdm_load_tile_bf16(A + (size_t)bm * K, lds_addr32(&As[0][0]), BM, BK, K, M);
        tdm_load_tile_bf16(Bt + (size_t)bn * K, lds_addr32(&Bs[0][0]), BN, BK, K, N);
    }
    for (int i = 0; i < niter; ++i) {
        int cur = i & 1;
        if (wave == 0) {
            if (i + 1 < niter) {
                int k0n = (i + 1) * BK;
                tdm_load_tile_bf16(A + (size_t)bm * K + k0n, lds_addr32(&As[cur ^ 1][0]), BM, BK, K, M);
                tdm_load_tile_bf16(Bt + (size_t)bn * K + k0n, lds_addr32(&Bs[cur ^ 1][0]), BN, BK, K, N);
                __builtin_amdgcn_s_wait_tensorcnt((short)2);  // current pair complete (in-order)
            } else {
                __builtin_amdgcn_s_wait_tensorcnt((short)0);
            }
        }
        __syncthreads();
        v16bf a = frag_rowmajor(&As[cur][0], wave * 16, BK, 0, lane);
#pragma unroll
        for (int t = 0; t < 4; ++t) {
            v16bf b = frag_rowmajor(&Bs[cur][0], t * 16, BK, 0, lane);
            acc[t] = wmma_bf16(a, b, acc[t]);
        }
        __syncthreads();  // tile consumed before wave0 overwrites it next iteration
    }
#else
    for (int i = 0; i < niter; ++i) {
        int k0 = i * BK;
        __syncthreads();
        {   // A tile 128x32
            int row = tid >> 1, hh = tid & 1;
            const unsigned int* src = (const unsigned int*)(A + (size_t)(bm + row) * K + k0 + hh * 16);
            unsigned int* dst = (unsigned int*)(&As[0][0] + row * BK + hh * 16);
#pragma unroll
            for (int j = 0; j < 8; ++j) dst[j] = src[j];
            if (k0 + BK < K) __builtin_prefetch(src + (BK >> 1), 0, 1);
        }
        {   // Bt tile 64x32
            int row = tid >> 2, qq = tid & 3;
            const unsigned int* src = (const unsigned int*)(Bt + (size_t)(bn + row) * K + k0 + qq * 8);
            unsigned int* dst = (unsigned int*)(&Bs[0][0] + row * BK + qq * 8);
#pragma unroll
            for (int j = 0; j < 4; ++j) dst[j] = src[j];
        }
        __syncthreads();
        v16bf a = frag_rowmajor(&As[0][0], wave * 16, BK, 0, lane);
#pragma unroll
        for (int t = 0; t < 4; ++t) {
            v16bf b = frag_rowmajor(&Bs[0][0], t * 16, BK, 0, lane);
            acc[t] = wmma_bf16(a, b, acc[t]);
        }
    }
#endif
    // epilogue (C layout: n = lane&15, m = r + 8*(lane>>4))
    int half = lane >> 4, nl = lane & 15;
#pragma unroll
    for (int t = 0; t < 4; ++t) {
        int gm = bm + wave * 16 + 8 * half;
        int gn = bn + t * 16 + nl;
#pragma unroll
        for (int r = 0; r < 8; ++r) {
            float v = acc[t][r];
            size_t off = (size_t)(gm + r) * N + gn;
            if (EPI == 0)      ((unsigned short*)Cout)[off] = f2bf(v);
            else if (EPI == 1) ((float*)Cout)[off] = v;
            else               ((float*)Cout)[off] = v + Res[off];
        }
    }
}

// ---------- short conv (+SiLU) for v ----------
__global__ void conv_silu_kernel(const unsigned short* __restrict__ lin,
                                 const float* __restrict__ cw,
                                 unsigned short* __restrict__ out) {
    size_t idx = (size_t)blockIdx.x * 256 + threadIdx.x;
    if (idx >= (size_t)B_ * T_ * D_) return;
    int d = (int)(idx % D_);
    size_t bt = idx / D_;
    int t = (int)(bt % T_);
    float y = 0.0f;
#pragma unroll
    for (int kk = 0; kk < 4; ++kk) {
        int tt = t - 3 + kk;
        if (tt >= 0) y += bf2f(lin[(bt + kk - 3) * D_ + d]) * cw[d * 4 + kk];
    }
    out[idx] = f2bf(silu(y));
}

// ---------- short conv + SiLU + RoPE for q/k ----------
__global__ void conv_silu_rope_kernel(const unsigned short* __restrict__ lin,
                                      const float* __restrict__ cw,
                                      unsigned short* __restrict__ out) {
    size_t idx = (size_t)blockIdx.x * 256 + threadIdx.x;
    if (idx >= (size_t)B_ * T_ * H_ * 64) return;
    int i64 = (int)(idx & 63);
    int h = (int)((idx >> 6) & (H_ - 1));
    size_t bt = idx >> 9;           // b*T + t
    int t = (int)(bt % T_);
    int d1 = h * DK_ + i64, d2 = d1 + 64;
    float y1 = 0.0f, y2 = 0.0f;
#pragma unroll
    for (int kk = 0; kk < 4; ++kk) {
        int tt = t - 3 + kk;
        if (tt >= 0) {
            size_t o = (bt + kk - 3) * D_;
            y1 += bf2f(lin[o + d1]) * cw[d1 * 4 + kk];
            y2 += bf2f(lin[o + d2]) * cw[d2 * 4 + kk];
        }
    }
    y1 = silu(y1);
    y2 = silu(y2);
    float invf = __powf(10000.0f, -(float)i64 / 64.0f);
    float f = (float)t * invf;
    float s, c;
    __sincosf(f, &s, &c);
    out[bt * D_ + d1] = f2bf(y1 * c - y2 * s);
    out[bt * D_ + d2] = f2bf(y1 * s + y2 * c);
}

// ---------- retention: one block per (b,h); 8 waves; S kept in WMMA accumulators ----------
__global__ __launch_bounds__(256) void retention_kernel(const unsigned short* __restrict__ q,
                                                        const unsigned short* __restrict__ k,
                                                        const unsigned short* __restrict__ v,
                                                        unsigned short* __restrict__ out_o) {
    extern __shared__ unsigned short smem[];
    unsigned short* qs = smem;                 // [64][128] q (unscaled; DK^-0.5 folded into epilogues)
    unsigned short* kb = qs + 64 * 128;        // [64][128]
    unsigned short* vb = kb + 64 * 128;        // [64][128]
    unsigned short* at = vb + 64 * 128;        // [64][64] masked att (bf16)
    unsigned short* st = at + 64 * 64;         // [128][128] S^T bf16: st[e*128+d]

    int tid = threadIdx.x, lane = tid & 31, wave = tid >> 5;
    int b = blockIdx.x >> 3, h = blockIdx.x & (H_ - 1);
    float gamma = 1.0f - exp2f(-5.0f - (float)h);
    float lg = __logf(gamma);
    float cdec = __expf(lg * (float)C_);
    const float scale = 0.08838834764831845f;  // DK^-0.5
    int nl = lane & 15, half = lane >> 4;

    for (int i = tid; i < 128 * 128; i += 256) st[i] = 0;

    v8f S[8];                                   // wave owns d-strip [wave*16, wave*16+16), all 8 e-tiles
#pragma unroll
    for (int e = 0; e < 8; ++e) S[e] = vzero8();

    for (int c = 0; c < NC_; ++c) {
        __syncthreads();
#if USE_TDM
        if (wave == 0) {
            size_t goff = ((size_t)b * T_ + (size_t)c * C_) * D_ + h * DK_;
            tdm_load_tile_bf16(q + goff, lds_addr32(qs), C_, 128, D_, (long long)B_ * T_);
            tdm_load_tile_bf16(k + goff, lds_addr32(kb), C_, 128, D_, (long long)B_ * T_);
            tdm_load_tile_bf16(v + goff, lds_addr32(vb), C_, 128, D_, (long long)B_ * T_);
            __builtin_amdgcn_s_wait_tensorcnt((short)0);
        }
#else
        {   // load chunk tiles (plain copies)
            int row = tid >> 2, qq = tid & 3;
            size_t gbase = ((size_t)b * T_ + (size_t)c * C_ + row) * D_ + h * DK_ + qq * 32;
            int lbase = row * 128 + qq * 32;
            const unsigned int* qsrc = (const unsigned int*)(q + gbase);
            const unsigned int* ksrc = (const unsigned int*)(k + gbase);
            const unsigned int* vsrc = (const unsigned int*)(v + gbase);
            unsigned int* qdst = (unsigned int*)(qs + lbase);
            unsigned int* kdst = (unsigned int*)(kb + lbase);
            unsigned int* vdst = (unsigned int*)(vb + lbase);
#pragma unroll
            for (int i = 0; i < 16; ++i) {
                qdst[i] = qsrc[i];
                kdst[i] = ksrc[i];
                vdst[i] = vsrc[i];
            }
        }
#endif
        __syncthreads();

        // ---- att = (q @ k^T) * scale * mask ----
#pragma unroll
        for (int it = 0; it < 2; ++it) {
            int tIdx = wave * 2 + it;
            int mi = tIdx >> 2, ni = tIdx & 3;
            v8f acc = vzero8();
#pragma unroll
            for (int ks = 0; ks < 4; ++ks) {
                v16bf a = frag_rowmajor(qs, mi * 16, 128, ks * 32, lane);
                v16bf bb = frag_rowmajor(kb, ni * 16, 128, ks * 32, lane);  // B^T = kb rows
                acc = wmma_bf16(a, bb, acc);
            }
            int j = ni * 16 + nl;
#pragma unroll
            for (int r = 0; r < 8; ++r) {
                int i = mi * 16 + r + 8 * half;
                int expo = i - j;
                float vv = (expo >= 0) ? acc[r] * (scale * __expf(lg * (float)expo)) : 0.0f;
                at[i * 64 + j] = f2bf(vv);
            }
        }
        __syncthreads();

        // ---- o = att@vb + (q*scale*cross)@S ----
#pragma unroll
        for (int it = 0; it < 4; ++it) {
            int tIdx = wave * 4 + it;
            int mi = tIdx >> 3, ni = tIdx & 7;
            v8f acc = vzero8();
#pragma unroll
            for (int ks = 0; ks < 2; ++ks) {
                v16bf a = frag_rowmajor(at, mi * 16, 64, ks * 32, lane);
                v16bf bb = frag_colstride(vb, ni * 16, 128, ks * 32, lane);  // B^T of vb
                acc = wmma_bf16(a, bb, acc);
            }
            float crossf = scale * __expf(lg * ((float)(mi * 16 + nl) + 1.0f));
#pragma unroll
            for (int ks = 0; ks < 4; ++ks) {
                v16bf a = frag_rowmajor_scaled(qs, mi * 16, 128, ks * 32, lane, crossf);
                v16bf bb = frag_rowmajor(st, ni * 16, 128, ks * 32, lane);   // B^T of S = S^T rows
                acc = wmma_bf16(a, bb, acc);
            }
            int e = ni * 16 + nl;
#pragma unroll
            for (int r = 0; r < 8; ++r) {
                int i = mi * 16 + r + 8 * half;
                out_o[((size_t)b * T_ + (size_t)c * C_ + i) * D_ + h * DV_ + e] = f2bf(acc[r]);
            }
        }
        __syncthreads();

        // ---- S = S*cdec + (k*kdec)^T @ v ----
#pragma unroll
        for (int ei = 0; ei < 8; ++ei) {
            v8f acc;
#pragma unroll
            for (int r = 0; r < 8; ++r) acc[r] = S[ei][r] * cdec;
#pragma unroll
            for (int ks = 0; ks < 2; ++ks) {
                v16bf a = frag_kT_dec(kb, wave * 16, ks * 32, lane, lg);
                v16bf bb = frag_colstride(vb, ei * 16, 128, ks * 32, lane);
                acc = wmma_bf16(a, bb, acc);
            }
            S[ei] = acc;
        }
        // mirror to LDS as bf16 S^T for next chunk's cross term
#pragma unroll
        for (int ei = 0; ei < 8; ++ei) {
            int e = ei * 16 + nl;
#pragma unroll
            for (int r = 0; r < 8; ++r) {
                int d = wave * 16 + r + 8 * half;
                st[e * 128 + d] = f2bf(S[ei][r]);
            }
        }
    }
}

// ---------- gated group-RMSNorm: out = rmsnorm(o, gnorm) * silu(g) ----------
__global__ __launch_bounds__(128) void gate_kernel(const unsigned short* __restrict__ o,
                                                   const unsigned short* __restrict__ g,
                                                   const float* __restrict__ gw,
                                                   unsigned short* __restrict__ out) {
    __shared__ float red[128];
    size_t base = (size_t)blockIdx.x * DV_;
    int i = threadIdx.x;
    float ov = bf2f(o[base + i]);
    red[i] = ov * ov;
    __syncthreads();
    for (int off = 64; off > 0; off >>= 1) {
        if (i < off) red[i] += red[i + off];
        __syncthreads();
    }
    float inv = rsqrtf(red[0] / (float)DV_ + GATE_EPS);
    float gv = bf2f(g[base + i]);
    out[base + i] = f2bf(ov * inv * gw[i] * silu(gv));
}

// ---------- m = silu(gate) * up ----------
__global__ void silu_mul_kernel(const unsigned short* __restrict__ gt,
                                const unsigned short* __restrict__ up,
                                unsigned short* __restrict__ out, size_t n) {
    size_t i = (size_t)blockIdx.x * 256 + threadIdx.x;
    if (i >= n) return;
    float g = bf2f(gt[i]);
    float u = bf2f(up[i]);
    out[i] = f2bf(silu(g) * u);
}

// ---------- host orchestration ----------
extern "C" void kernel_launch(void* const* d_in, const int* in_sizes, int n_in,
                              void* d_out, int out_size, void* d_ws, size_t ws_size,
                              hipStream_t stream) {
    const float* hidden = (const float*)d_in[0];
    const float* ln1_w = (const float*)d_in[1];
    const float* ln2_w = (const float*)d_in[2];
    const float* wq = (const float*)d_in[3];
    const float* wk = (const float*)d_in[4];
    const float* wv = (const float*)d_in[5];
    const float* wg = (const float*)d_in[6];
    const float* wo = (const float*)d_in[7];
    const float* conv_q_w = (const float*)d_in[8];
    const float* conv_k_w = (const float*)d_in[9];
    const float* conv_v_w = (const float*)d_in[10];
    const float* gnorm_w = (const float*)d_in[11];
    const float* w_gate = (const float*)d_in[12];
    const float* w_up = (const float*)d_in[13];
    const float* w_down = (const float*)d_in[14];
    float* out = (float*)d_out;

    const size_t BT = (size_t)B_ * T_;
    char* ws = (char*)d_ws;
    size_t off = 0;
    auto alloc = [&](size_t elems) -> unsigned short* {
        unsigned short* p = (unsigned short*)(ws + off);
        off += ((elems * 2 + 255) & ~(size_t)255);
        return p;
    };
    unsigned short* wq_t = alloc((size_t)D_ * D_);
    unsigned short* wk_t = alloc((size_t)D_ * D_);
    unsigned short* wv_t = alloc((size_t)D_ * D_);
    unsigned short* wg_t = alloc((size_t)D_ * D_);
    unsigned short* wo_t = alloc((size_t)D_ * D_);
    unsigned short* wgate_t = alloc((size_t)I_ * D_);   // [I][D]
    unsigned short* wup_t = alloc((size_t)I_ * D_);     // [I][D]
    unsigned short* wdown_t = alloc((size_t)D_ * I_);   // [D][I]
    unsigned short* xb = alloc(BT * D_);
    unsigned short* qlin = alloc(BT * D_);
    unsigned short* klin = alloc(BT * D_);
    unsigned short* vlin = alloc(BT * D_);
    unsigned short* qb = alloc(BT * D_);                // q,k,v contiguous -> reused for 'gate'
    unsigned short* kbuf = alloc(BT * D_);
    unsigned short* vbuf = alloc(BT * D_);
    unsigned short* gb = alloc(BT * D_);
    unsigned short* upb = alloc(BT * (size_t)I_);
    unsigned short* mb = alloc(BT * (size_t)I_);
    // reuse dead buffers
    unsigned short* ob = qlin;     // retention output (qlin dead after conv)
    unsigned short* ogb = klin;    // gated output
    unsigned short* yb = vlin;     // rmsnorm2 output
    unsigned short* gateb = qb;    // MLP gate proj (q/k/v dead after retention)

    const dim3 blk256(256);

    // 1) weight transpose+cast
    transpose_cast_kernel<<<(unsigned)(((size_t)D_ * D_ + 255) / 256), blk256, 0, stream>>>(wq, wq_t, D_, D_);
    transpose_cast_kernel<<<(unsigned)(((size_t)D_ * D_ + 255) / 256), blk256, 0, stream>>>(wk, wk_t, D_, D_);
    transpose_cast_kernel<<<(unsigned)(((size_t)D_ * D_ + 255) / 256), blk256, 0, stream>>>(wv, wv_t, D_, D_);
    transpose_cast_kernel<<<(unsigned)(((size_t)D_ * D_ + 255) / 256), blk256, 0, stream>>>(wg, wg_t, D_, D_);
    transpose_cast_kernel<<<(unsigned)(((size_t)D_ * D_ + 255) / 256), blk256, 0, stream>>>(wo, wo_t, D_, D_);
    transpose_cast_kernel<<<(unsigned)(((size_t)D_ * I_ + 255) / 256), blk256, 0, stream>>>(w_gate, wgate_t, D_, I_);
    transpose_cast_kernel<<<(unsigned)(((size_t)D_ * I_ + 255) / 256), blk256, 0, stream>>>(w_up, wup_t, D_, I_);
    transpose_cast_kernel<<<(unsigned)(((size_t)I_ * D_ + 255) / 256), blk256, 0, stream>>>(w_down, wdown_t, I_, D_);

    // 2) x = rmsnorm(hidden, ln1)
    rmsnorm_kernel<<<(unsigned)BT, blk256, 0, stream>>>(hidden, ln1_w, xb, RMS_EPS);

    // 3) projections (bf16 WMMA GEMMs, TDM-fed)
    dim3 gDD(D_ / BN, (unsigned)(BT / BM));
    gemm_bf16<0><<<gDD, blk256, 0, stream>>>(xb, wq_t, qlin, nullptr, (int)BT, D_, D_);
    gemm_bf16<0><<<gDD, blk256, 0, stream>>>(xb, wk_t, klin, nullptr, (int)BT, D_, D_);
    gemm_bf16<0><<<gDD, blk256, 0, stream>>>(xb, wv_t, vlin, nullptr, (int)BT, D_, D_);
    gemm_bf16<0><<<gDD, blk256, 0, stream>>>(xb, wg_t, gb, nullptr, (int)BT, D_, D_);

    // 4) short conv + silu (+rope)
    conv_silu_rope_kernel<<<(unsigned)((BT * H_ * 64 + 255) / 256), blk256, 0, stream>>>(qlin, conv_q_w, qb);
    conv_silu_rope_kernel<<<(unsigned)((BT * H_ * 64 + 255) / 256), blk256, 0, stream>>>(klin, conv_k_w, kbuf);
    conv_silu_kernel<<<(unsigned)((BT * D_ + 255) / 256), blk256, 0, stream>>>(vlin, conv_v_w, vbuf);

    // 5) retention (88 KB dynamic LDS)
    size_t ret_lds = (size_t)(3 * 64 * 128 + 64 * 64 + 128 * 128) * sizeof(unsigned short);
    retention_kernel<<<dim3(B_ * H_), blk256, ret_lds, stream>>>(qb, kbuf, vbuf, ob);

    // 6) gated group norm
    gate_kernel<<<(unsigned)(BT * H_), dim3(128), 0, stream>>>(ob, gb, gnorm_w, ogb);

    // 7) h = hidden + og @ wo   (f32 into d_out)
    gemm_bf16<2><<<gDD, blk256, 0, stream>>>(ogb, wo_t, out, hidden, (int)BT, D_, D_);

    // 8) y = rmsnorm(h, ln2)
    rmsnorm_kernel<<<(unsigned)BT, blk256, 0, stream>>>(out, ln2_w, yb, RMS_EPS);

    // 9) MLP up/gate
    dim3 gDI(I_ / BN, (unsigned)(BT / BM));
    gemm_bf16<0><<<gDI, blk256, 0, stream>>>(yb, wgate_t, gateb, nullptr, (int)BT, I_, D_);
    gemm_bf16<0><<<gDI, blk256, 0, stream>>>(yb, wup_t, upb, nullptr, (int)BT, I_, D_);

    // 10) m = silu(gate) * up
    silu_mul_kernel<<<(unsigned)((BT * I_ + 255) / 256), blk256, 0, stream>>>(gateb, upb, mb, BT * (size_t)I_);

    // 11) out = h + m @ w_down  (residual read+write d_out elementwise, safe)
    gemm_bf16<2><<<gDD, blk256, 0, stream>>>(mb, wdown_t, out, out, (int)BT, D_, I_);

    (void)in_sizes; (void)n_in; (void)out_size; (void)ws_size;
}
